// Attention_10084583211652
// MI455X (gfx1250) — matrix operand order repaired and checked by reading, not compiled, not run on hardware
//
#include <hip/hip_runtime.h>
#include <cstdint>
#include <cstddef>

#define BATCH 2
#define SEQ   2048
#define DMODEL 2048
#define HEADS 16
#define DH    128
#define MROWS (BATCH * SEQ)   // 4096

typedef __attribute__((ext_vector_type(16))) __bf16 v16bf;
typedef __attribute__((ext_vector_type(8)))  __bf16 v8bf;
typedef __attribute__((ext_vector_type(4)))  __bf16 v4bf;
typedef __attribute__((ext_vector_type(8)))  float  v8f;

union ABu { v16bf v; v8bf h[2]; };

__device__ __forceinline__ __bf16 f2bf(float f) {
    union { float f; unsigned u; } c; c.f = f;
    unsigned r = c.u + 0x7FFFu + ((c.u >> 16) & 1u);   // round-to-nearest-even
    unsigned short hs = (unsigned short)(r >> 16);
    return __builtin_bit_cast(__bf16, hs);
}

__device__ __forceinline__ v8f wmma_bf16(v16bf a, v16bf b, v8f c) {
    return __builtin_amdgcn_wmma_f32_16x16x32_bf16(
        /*neg_a=*/false, a, /*neg_b=*/false, b,
        /*c_mod=*/(short)0, c, /*reuse_a=*/false, /*reuse_b=*/false);
}

// ds_swizzle XOR butterfly within 16-lane halves (masks 1,2,4,8 stay in-half).
template <int OFF>
__device__ __forceinline__ float swz(float v) {
    return __builtin_bit_cast(float,
        __builtin_amdgcn_ds_swizzle(__builtin_bit_cast(int, v), OFF));
}

// gfx1250 async global->LDS copy, 16 bytes per lane (ASYNCcnt-tracked).
// GVS addressing: saddr(SGPR64 base) + vaddr(per-lane u32 byte offset).
__device__ __forceinline__ void async_b128(const void* gbase, unsigned gbyte,
                                           unsigned lbyte) {
    asm volatile("global_load_async_to_lds_b128 %0, %1, %2"
                 :: "v"(lbyte), "v"(gbyte), "s"(gbase)
                 : "memory");
}

__device__ __forceinline__ void wait_async_le4() {
    asm volatile("s_wait_asynccnt 4" ::: "memory");
}
__device__ __forceinline__ void wait_async_le0() {
    asm volatile("s_wait_asynccnt 0" ::: "memory");
}

// ---------------------------------------------------------------------------
// fp32 -> bf16 elementwise conversion (vectorized 4-wide)
// ---------------------------------------------------------------------------
__global__ void cvt_f32_bf16(const float* __restrict__ in,
                             __bf16* __restrict__ out, int n4) {
    int i = blockIdx.x * blockDim.x + threadIdx.x;
    if (i < n4) {
        float4 f = ((const float4*)in)[i];
        v4bf o;
        o[0] = f2bf(f.x); o[1] = f2bf(f.y); o[2] = f2bf(f.z); o[3] = f2bf(f.w);
        ((v4bf*)out)[i] = o;
    }
}

// ---------------------------------------------------------------------------
// bf16 WMMA GEMM with async-LDS double buffering:
//   C[M,N] = A[M,K] @ W[N,K]^T + bias
// Block = 4 waves = 64 rows x 64 cols. Per k-step(32) the block stages a
// 64x32 A tile and a 64x32 B tile (4 KiB each) into LDS with
// global_load_async_to_lds_b128, double-buffered so tile k+1 streams in
// while WMMAs consume tile k from LDS.
// MODE 0: bf16 row-major [M,N]     MODE 1: bf16 [b,h,s,dh] (Q/K layout)
// MODE 2: bf16 [b,h,dh,s] (V^T)    MODE 3: f32 row-major [M,N]
// ---------------------------------------------------------------------------
template <int MODE>
__global__ __launch_bounds__(128)
void gemm_bf16_wmma(const __bf16* __restrict__ A, const __bf16* __restrict__ W,
                    const float* __restrict__ bias, void* __restrict__ Cout,
                    int M, int N, int K) {
    const int tid  = threadIdx.x;
    const int lane = tid & 31;
    const int wave = tid >> 5;
    const int half = lane >> 4;
    const int l16  = lane & 15;
    const int row0b = blockIdx.y * 64;          // block's first row
    const int row0  = row0b + wave * 16;        // wave's first row
    const int col0  = blockIdx.x * 64;

    __shared__ __bf16 sA[2][64 * 32];           // [row][k] 4 KiB per buffer
    __shared__ __bf16 sB[2][64 * 32];           // [col][k] 4 KiB per buffer

    // Stage one 64x32 A tile + 64x32 B tile into LDS buffer `buf`.
    // 256 16-byte chunks per tile, 128 threads -> 2 chunks each -> 4 async
    // instructions per thread per buffer (ASYNCcnt += 4).
    auto issue = [&](int buf, int k0) {
#pragma unroll
        for (int i = 0; i < 2; ++i) {
            const int c   = tid + i * 128;      // 0..255
            const int r   = c >> 2;             // row/col within tile
            const int q   = c & 3;              // 16B chunk within 64B k-row
            const unsigned ga = (unsigned)(((row0b + r) * K + k0 + q * 8) * 2);
            const unsigned la = (unsigned)(size_t)(&sA[buf][r * 32 + q * 8]);
            async_b128(A, ga, la);
            const unsigned gb = (unsigned)(((col0 + r) * K + k0 + q * 8) * 2);
            const unsigned lb = (unsigned)(size_t)(&sB[buf][r * 32 + q * 8]);
            async_b128(W, gb, lb);
        }
    };

    v8f acc[4] = {};
    issue(0, 0);                                // prologue: fill buffer 0

    for (int k0 = 0; k0 < K; k0 += 32) {
        const int cur = (k0 >> 5) & 1;
        const bool more = (k0 + 32) < K;
        if (more) issue(cur ^ 1, k0 + 32);      // overlap next tile's copy
        if (more) wait_async_le4();             // current tile's 4 ops done
        else      wait_async_le0();
        __syncthreads();                        // tile visible block-wide

        ABu a;
        const __bf16* sAp = &sA[cur][(wave * 16 + l16) * 32];
        a.v = *(const v16bf*)(sAp + half * 16);
#pragma unroll
        for (int n = 0; n < 4; ++n) {
            const v16bf b = *(const v16bf*)(&sB[cur][(n * 16 + l16) * 32 + half * 16]);
            acc[n] = wmma_bf16(a.v, b, acc[n]);
        }
        __syncthreads();                        // reads done before overwrite
    }

#pragma unroll
    for (int n = 0; n < 4; ++n) {
        const int col = col0 + n * 16 + l16;
        const float bv = bias[col];
#pragma unroll
        for (int r = 0; r < 8; ++r) {
            const int row = row0 + half * 8 + r;
            const float val = acc[n][r] + bv;
            if (MODE == 0) {
                ((__bf16*)Cout)[(size_t)row * N + col] = f2bf(val);
            } else if (MODE == 1) {           // [b,h,s,dh]
                int bb = row / SEQ, s = row % SEQ, hh = col / DH, dd = col % DH;
                ((__bf16*)Cout)[(((size_t)(bb * HEADS + hh)) * SEQ + s) * DH + dd] = f2bf(val);
            } else if (MODE == 2) {           // [b,h,dh,s]
                int bb = row / SEQ, s = row % SEQ, hh = col / DH, dd = col % DH;
                ((__bf16*)Cout)[(((size_t)(bb * HEADS + hh)) * DH + dd) * SEQ + s] = f2bf(val);
            } else {                          // f32 [M,N]
                ((float*)Cout)[(size_t)row * N + col] = val;
            }
        }
    }
}

// ---------------------------------------------------------------------------
// Flash attention: one wave per (b, h, 16-query tile).
// Q,K: [b,h,s,dh] bf16.  Vt: [b,h,dh,s] bf16.  Out: [b*s, h*dh] bf16.
// ---------------------------------------------------------------------------
__global__ __launch_bounds__(32)
void flash_attn_bf16(const __bf16* __restrict__ Qb, const __bf16* __restrict__ Kb,
                     const __bf16* __restrict__ Vt, __bf16* __restrict__ Ob) {
    const int lane = threadIdx.x & 31;
    const int half = lane >> 4;
    const int l16  = lane & 15;

    int t = blockIdx.x;
    const int qt = t % (SEQ / 16); t /= (SEQ / 16);
    const int h  = t % HEADS;
    const int b  = t / HEADS;
    const int q0 = qt * 16;
    const size_t bh = (size_t)(b * HEADS + h);

    __shared__ __bf16 sP[16 * 32];

    // Q fragments for the whole dh=128, kept in registers (4 K-chunks of 32).
    ABu qf[4];
    const __bf16* qbase = Qb + (bh * SEQ + q0 + l16) * DH;
#pragma unroll
    for (int kc = 0; kc < 4; ++kc) {
        qf[kc].v = *(const v16bf*)(qbase + kc * 32 + half * 16);
    }

    v8f o[8] = {};
    float mrow[8], lrow[8];
#pragma unroll
    for (int r = 0; r < 8; ++r) { mrow[r] = -1e30f; lrow[r] = 0.0f; }
    const float scale = 0.08838834764831845f;   // 1/sqrt(128)

    for (int kb = 0; kb < SEQ; kb += 32) {
        // ---- scores: 16x32 tile = 2 accumulators, K-dim 128 = 4 WMMAs each
        v8f s0 = {}, s1 = {};
#pragma unroll
        for (int kc = 0; kc < 4; ++kc) {
            const __bf16* kp0 = Kb + (bh * SEQ + kb + l16) * DH + kc * 32 + half * 16;
            const __bf16* kp1 = kp0 + (size_t)16 * DH;
            s0 = wmma_bf16(qf[kc].v, *(const v16bf*)kp0, s0);
            s1 = wmma_bf16(qf[kc].v, *(const v16bf*)kp1, s1);
        }

        // ---- online softmax (row = half*8 + r lives across the 16-lane half)
        float corr[8];
#pragma unroll
        for (int r = 0; r < 8; ++r) {
            float a0 = s0[r] * scale, a1 = s1[r] * scale;
            float mx = fmaxf(a0, a1);
            mx = fmaxf(mx, swz<0x041F>(mx));
            mx = fmaxf(mx, swz<0x081F>(mx));
            mx = fmaxf(mx, swz<0x101F>(mx));
            mx = fmaxf(mx, swz<0x201F>(mx));
            const float mnew = fmaxf(mrow[r], mx);
            corr[r] = __expf(mrow[r] - mnew);
            mrow[r] = mnew;
            const float p0 = __expf(a0 - mnew);
            const float p1 = __expf(a1 - mnew);
            float ps = p0 + p1;
            ps += swz<0x041F>(ps);
            ps += swz<0x081F>(ps);
            ps += swz<0x101F>(ps);
            ps += swz<0x201F>(ps);
            lrow[r] = lrow[r] * corr[r] + ps;
            const int m = half * 8 + r;           // C-layout -> LDS row-major
            sP[m * 32 + l16]      = f2bf(p0);
            sP[m * 32 + 16 + l16] = f2bf(p1);
        }
        __syncthreads();

        // ---- reload P in A-matrix layout
        ABu pf;
        pf.v = *(const v16bf*)(&sP[l16 * 32 + half * 16]);

        // ---- rescale running output, then accumulate P @ V
#pragma unroll
        for (int n = 0; n < 8; ++n) {
#pragma unroll
            for (int r = 0; r < 8; ++r) o[n][r] *= corr[r];
        }
#pragma unroll
        for (int n = 0; n < 8; ++n) {
            const __bf16* vp = Vt + (bh * DH + n * 16 + l16) * SEQ + kb + half * 16;
            o[n] = wmma_bf16(pf.v, *(const v16bf*)vp, o[n]);
        }
        __syncthreads();
    }

    // ---- normalize and store [b*s, h*dh]
#pragma unroll
    for (int r = 0; r < 8; ++r) lrow[r] = 1.0f / lrow[r];
#pragma unroll
    for (int n = 0; n < 8; ++n) {
#pragma unroll
        for (int r = 0; r < 8; ++r) {
            const int row = b * SEQ + q0 + half * 8 + r;
            const int col = h * DH + n * 16 + l16;
            Ob[(size_t)row * DMODEL + col] = f2bf(o[n][r] * lrow[r]);
        }
    }
}

// ---------------------------------------------------------------------------
// Host launcher
// ---------------------------------------------------------------------------
extern "C" void kernel_launch(void* const* d_in, const int* in_sizes, int n_in,
                              void* d_out, int out_size, void* d_ws, size_t ws_size,
                              hipStream_t stream) {
    const float* x  = (const float*)d_in[0];
    const float* wq = (const float*)d_in[1];
    const float* bq = (const float*)d_in[2];
    const float* wk = (const float*)d_in[3];
    const float* bk = (const float*)d_in[4];
    const float* wv = (const float*)d_in[5];
    const float* bv = (const float*)d_in[6];
    const float* wo = (const float*)d_in[7];
    const float* bo = (const float*)d_in[8];

    char* ws = (char*)d_ws;
    const size_t XB  = (size_t)MROWS * DMODEL * 2;        // 16 MiB
    const size_t WB  = (size_t)DMODEL * DMODEL * 2;       // 8 MiB
    __bf16* xb    = (__bf16*)(ws);
    __bf16* wqb   = (__bf16*)(ws + XB);
    __bf16* wkb   = (__bf16*)(ws + XB + WB);
    __bf16* wvb   = (__bf16*)(ws + XB + 2 * WB);
    __bf16* wob   = (__bf16*)(ws + XB + 3 * WB);
    __bf16* Qb    = (__bf16*)(ws + XB + 4 * WB);
    __bf16* Kb    = (__bf16*)(ws + 2 * XB + 4 * WB);
    __bf16* Vt    = (__bf16*)(ws + 3 * XB + 4 * WB);
    __bf16* attnb = (__bf16*)(ws + 4 * XB + 4 * WB);

    // 1) down-convert activations + weights to bf16
    {
        const int nx4 = MROWS * DMODEL / 4;
        const int nw4 = DMODEL * DMODEL / 4;
        cvt_f32_bf16<<<(nx4 + 255) / 256, 256, 0, stream>>>(x,  xb,  nx4);
        cvt_f32_bf16<<<(nw4 + 255) / 256, 256, 0, stream>>>(wq, wqb, nw4);
        cvt_f32_bf16<<<(nw4 + 255) / 256, 256, 0, stream>>>(wk, wkb, nw4);
        cvt_f32_bf16<<<(nw4 + 255) / 256, 256, 0, stream>>>(wv, wvb, nw4);
        cvt_f32_bf16<<<(nw4 + 255) / 256, 256, 0, stream>>>(wo, wob, nw4);
    }

    // 2) Q/K/V projections (bf16 WMMA, fp32 accumulate, async-LDS pipelined)
    dim3 ggrid(DMODEL / 64, MROWS / 64);   // (32, 64)
    gemm_bf16_wmma<1><<<ggrid, 128, 0, stream>>>(xb, wqb, bq, Qb, MROWS, DMODEL, DMODEL);
    gemm_bf16_wmma<1><<<ggrid, 128, 0, stream>>>(xb, wkb, bk, Kb, MROWS, DMODEL, DMODEL);
    gemm_bf16_wmma<2><<<ggrid, 128, 0, stream>>>(xb, wvb, bv, Vt, MROWS, DMODEL, DMODEL);

    // 3) flash attention: B*H*(S/16) = 4096 waves
    flash_attn_bf16<<<BATCH * HEADS * (SEQ / 16), 32, 0, stream>>>(Qb, Kb, Vt, attnb);

    // 4) output projection -> fp32 result
    gemm_bf16_wmma<3><<<ggrid, 128, 0, stream>>>(attnb, wob, bo, d_out, MROWS, DMODEL, DMODEL);
}
